// Triaffine_48576080118515
// MI455X (gfx1250) — matrix operand order repaired and checked by reading, not compiled, not run on hardware
//
#include <hip/hip_runtime.h>
#include <stdint.h>

// Triaffine: out[n,i,j,k] = sum_{a,b,c} l1[n,i,a] W[a,b,c] l2[n,j,b] l3[n,k,c]
// B=4, L=160, D=256, DP=257 (bias). All GEMMs run as bf16 WMMA (16x16x32, f32 acc),
// register-blocked so each wave computes multiple 16x16 tiles per K-step.

#define BATCH 4
#define LEN   160
#define DIN   256
#define DP    257
#define PD    288            // DP padded to a multiple of 32 (K granularity)
#define PD2   (PD * PD)      // 82944
#define M1    (BATCH * LEN)  // 640
#define LDSTRIDE 296         // LDS row stride (halves) for the stage-2 result

typedef __attribute__((ext_vector_type(16))) __bf16       v16bf;
typedef __attribute__((ext_vector_type(8)))  float        v8f;
typedef __attribute__((ext_vector_type(4)))  unsigned int v4u;

union Frag { v16bf v; v4u q[2]; };

__device__ __forceinline__ unsigned short f2bf(float f) {
  union { float f; unsigned int u; } c; c.f = f;
  unsigned int u = c.u;
  return (unsigned short)((u + 0x7FFFu + ((u >> 16) & 1u)) >> 16); // RNE
}

// Fragment layouts (wave32, 16-bit WMMA, per CDNA5 ISA 7.12.2):
//   A (16x32): lane L holds row M=L&15; halves h<8 -> K=(L<16?0:8)+h,
//              h>=8 -> K=16+(L<16?0:8)+(h-8)   => two 16B loads, 32B apart.
//   B (32x16): lane L holds col N=L&15; same K mapping as A
//              => two 16B loads, 32B apart.
//   C/D f32 16x16: lane<16: N=lane, rows M=r..r+7; lane>=16: N=lane-16, M=8+r.
__device__ __forceinline__ void load_afrag(Frag& f, const unsigned short* p) {
  const v4u* q = (const v4u*)p;
  f.q[0] = q[0]; f.q[1] = q[2];
}
__device__ __forceinline__ void load_bfrag(Frag& f, const unsigned short* p) {
  const v4u* q = (const v4u*)p;
  f.q[0] = q[0]; f.q[1] = q[2];
}

// ---- prep: pad [row, 0..255]=x, [row,256]=1 (bias), [row,257..287]=0, as bf16
__global__ void pad_bias_bf16(const float* __restrict__ src,
                              unsigned short* __restrict__ dst) {
  int idx = blockIdx.x * blockDim.x + threadIdx.x;
  if (idx >= M1 * PD) return;
  int c = idx % PD;
  int row = idx / PD;
  float v = 0.0f;
  if (c < DIN)       v = src[row * DIN + c];
  else if (c == DIN) v = 1.0f;           // bias element (a == 256)
  dst[idx] = f2bf(v);
}

// ---- prep: WbT[(c*PD + b)*PD + a] = W[a,b,c], zero-padded, as bf16
__global__ void w_transpose_bf16(const float* __restrict__ W,
                                 unsigned short* __restrict__ WbT) {
  int idx = blockIdx.x * blockDim.x + threadIdx.x;
  if (idx >= PD * PD2) return;
  int a = idx % PD;
  int q = idx / PD;
  int b = q % PD;
  int c = q / PD;
  float v = 0.0f;
  if (a < DP && b < DP && c < DP) v = W[((a * DP) + b) * DP + c];
  WbT[idx] = f2bf(v);
}

// ---- stage 1: T1[(n,i)][c*PD+b] = sum_a L1b[(n,i),a] * WbT[(c*PD+b),a]
// 8 waves/WG as 2(M)x4(N); each wave computes a 32x64 block (2x4 accumulators,
// 8 WMMAs per K-step on 6KB of operands => ~22 FLOP/B from L2).
__global__ __launch_bounds__(256) void triaffine_stage1(
    const unsigned short* __restrict__ A,   // [M1][PD]   row-major
    const unsigned short* __restrict__ Bt,  // [PD2][PD]  (B transposed)
    unsigned short* __restrict__ C)         // [M1][PD2]  bf16
{
  const int wave = threadIdx.x >> 5;
  const int lane = threadIdx.x & 31;
  const int lr   = lane & 15;
  const int hi   = (lane & 16) ? 1 : 0;
  const int akh  = hi * 8;
  const int bkh  = hi * 8;
  const int wm   = wave & 1;                // 2 waves along M
  const int wn   = wave >> 1;               // 4 waves along N
  const int Mb   = blockIdx.y * 64 + wm * 32;   // grid.y = M1/64 = 10
  const int Nb   = blockIdx.x * 256 + wn * 64;  // grid.x = PD2/256 = 324

  const unsigned short* arow[2];
  const unsigned short* brow[4];
#pragma unroll
  for (int mi = 0; mi < 2; ++mi)
    arow[mi] = A + (size_t)(Mb + mi * 16 + lr) * PD + akh;
#pragma unroll
  for (int ni = 0; ni < 4; ++ni)
    brow[ni] = Bt + (size_t)(Nb + ni * 16 + lr) * PD + bkh;

  v8f acc[2][4];
#pragma unroll
  for (int mi = 0; mi < 2; ++mi)
#pragma unroll
    for (int ni = 0; ni < 4; ++ni)
      acc[mi][ni] = (v8f){0.f, 0.f, 0.f, 0.f, 0.f, 0.f, 0.f, 0.f};

  for (int k0 = 0; k0 < PD; k0 += 32) {
    Frag aF[2], bF[4];
#pragma unroll
    for (int mi = 0; mi < 2; ++mi) load_afrag(aF[mi], arow[mi] + k0);
#pragma unroll
    for (int ni = 0; ni < 4; ++ni) load_bfrag(bF[ni], brow[ni] + k0);
#pragma unroll
    for (int mi = 0; mi < 2; ++mi)
#pragma unroll
      for (int ni = 0; ni < 4; ++ni)
        acc[mi][ni] = __builtin_amdgcn_wmma_f32_16x16x32_bf16(
            false, aF[mi].v, false, bF[ni].v, (short)0, acc[mi][ni], false, false);
  }

#pragma unroll
  for (int mi = 0; mi < 2; ++mi) {
    const int rb = Mb + mi * 16 + hi * 8;
#pragma unroll
    for (int ni = 0; ni < 4; ++ni) {
      const int col = Nb + ni * 16 + lr;
      unsigned short* crow = C + (size_t)rb * PD2 + col;
#pragma unroll
      for (int r = 0; r < 8; ++r) crow[(size_t)r * PD2] = f2bf(acc[mi][ni][r]);
    }
  }
}

// ---- fused stages 2+3: one workgroup per (n,i); P kept in LDS.
//  phase1: P[j][c] = sum_b L2b[n][j,b] * T1[(n,i)][c*PD+b]   (T1 row = Bt)
//  phase2: out[n,i,j,k] = sum_c P[j][c] * L3b[n][k,c]        (L3b row = Bt)
// Both phases use 32x32 register blocks (2x2 accumulators per wave).
__global__ __launch_bounds__(256) void triaffine_stage23(
    const unsigned short* __restrict__ L2b, // [BATCH][LEN][PD]
    const unsigned short* __restrict__ L3b, // [BATCH][LEN][PD]
    const unsigned short* __restrict__ T1,  // [M1][PD2], each row is [c][b]
    float* __restrict__ Out)                // [BATCH][LEN][LEN][LEN]
{
  extern __shared__ unsigned short Pl[];    // [LEN][LDSTRIDE] bf16
  const int t    = blockIdx.x;              // t = n*LEN + i
  const int n    = t / LEN;
  const int wave = threadIdx.x >> 5;
  const int lane = threadIdx.x & 31;
  const int lr   = lane & 15;
  const int hi   = (lane & 16) ? 1 : 0;
  const int akh  = hi * 8;
  const int bkh  = hi * 8;

  const unsigned short* A2 = L2b + (size_t)n * (LEN * PD);
  const unsigned short* B2 = T1 + (size_t)t * PD2;

  // ---- phase 1: 160x288 GEMM (K=288), result into LDS. 5x9 = 45 blocks.
  for (int blk = wave; blk < (LEN / 32) * (PD / 32); blk += 8) {
    const int Mb = (blk / (PD / 32)) * 32;
    const int Nb = (blk % (PD / 32)) * 32;
    const unsigned short* arow[2];
    const unsigned short* brow[2];
#pragma unroll
    for (int mi = 0; mi < 2; ++mi)
      arow[mi] = A2 + (size_t)(Mb + mi * 16 + lr) * PD + akh;
#pragma unroll
    for (int ni = 0; ni < 2; ++ni)
      brow[ni] = B2 + (size_t)(Nb + ni * 16 + lr) * PD + bkh;

    v8f acc[2][2];
#pragma unroll
    for (int mi = 0; mi < 2; ++mi)
#pragma unroll
      for (int ni = 0; ni < 2; ++ni)
        acc[mi][ni] = (v8f){0.f, 0.f, 0.f, 0.f, 0.f, 0.f, 0.f, 0.f};

    for (int k0 = 0; k0 < PD; k0 += 32) {
      Frag aF[2], bF[2];
#pragma unroll
      for (int mi = 0; mi < 2; ++mi) load_afrag(aF[mi], arow[mi] + k0);
#pragma unroll
      for (int ni = 0; ni < 2; ++ni) load_bfrag(bF[ni], brow[ni] + k0);
#pragma unroll
      for (int mi = 0; mi < 2; ++mi)
#pragma unroll
        for (int ni = 0; ni < 2; ++ni)
          acc[mi][ni] = __builtin_amdgcn_wmma_f32_16x16x32_bf16(
              false, aF[mi].v, false, bF[ni].v, (short)0, acc[mi][ni], false, false);
    }

#pragma unroll
    for (int mi = 0; mi < 2; ++mi) {
      const int rb = Mb + mi * 16 + hi * 8;
#pragma unroll
      for (int ni = 0; ni < 2; ++ni) {
        const int col = Nb + ni * 16 + lr;
#pragma unroll
        for (int r = 0; r < 8; ++r)
          Pl[(rb + r) * LDSTRIDE + col] = f2bf(acc[mi][ni][r]);
      }
    }
  }

  __syncthreads();

  // ---- phase 2: 160x160 GEMM (A from LDS, K=288), f32 straight to out. 25 blocks.
  const unsigned short* B3 = L3b + (size_t)n * (LEN * PD);
  float* O = Out + (size_t)t * (LEN * LEN);
  for (int blk = wave; blk < (LEN / 32) * (LEN / 32); blk += 8) {
    const int Mb = (blk / (LEN / 32)) * 32;
    const int Nb = (blk % (LEN / 32)) * 32;
    const unsigned short* arow[2];
    const unsigned short* brow[2];
#pragma unroll
    for (int mi = 0; mi < 2; ++mi)
      arow[mi] = Pl + (Mb + mi * 16 + lr) * LDSTRIDE + akh;
#pragma unroll
    for (int ni = 0; ni < 2; ++ni)
      brow[ni] = B3 + (size_t)(Nb + ni * 16 + lr) * PD + bkh;

    v8f acc[2][2];
#pragma unroll
    for (int mi = 0; mi < 2; ++mi)
#pragma unroll
      for (int ni = 0; ni < 2; ++ni)
        acc[mi][ni] = (v8f){0.f, 0.f, 0.f, 0.f, 0.f, 0.f, 0.f, 0.f};

    for (int k0 = 0; k0 < PD; k0 += 32) {
      Frag aF[2], bF[2];
#pragma unroll
      for (int mi = 0; mi < 2; ++mi) load_afrag(aF[mi], arow[mi] + k0);
#pragma unroll
      for (int ni = 0; ni < 2; ++ni) load_bfrag(bF[ni], brow[ni] + k0);
#pragma unroll
      for (int mi = 0; mi < 2; ++mi)
#pragma unroll
        for (int ni = 0; ni < 2; ++ni)
          acc[mi][ni] = __builtin_amdgcn_wmma_f32_16x16x32_bf16(
              false, aF[mi].v, false, bF[ni].v, (short)0, acc[mi][ni], false, false);
    }

#pragma unroll
    for (int mi = 0; mi < 2; ++mi) {
      const int rb = Mb + mi * 16 + hi * 8;
#pragma unroll
      for (int ni = 0; ni < 2; ++ni) {
        const int col = Nb + ni * 16 + lr;
#pragma unroll
        for (int r = 0; r < 8; ++r) O[(rb + r) * LEN + col] = acc[mi][ni][r];
      }
    }
  }
}

extern "C" void kernel_launch(void* const* d_in, const int* in_sizes, int n_in,
                              void* d_out, int out_size, void* d_ws, size_t ws_size,
                              hipStream_t stream) {
  (void)in_sizes; (void)n_in; (void)out_size;
  const float* l1 = (const float*)d_in[0];
  const float* l2 = (const float*)d_in[1];
  const float* l3 = (const float*)d_in[2];
  const float* W  = (const float*)d_in[3];
  float* out = (float*)d_out;

  char* ws = (char*)d_ws;
  const size_t SZ_L  = (size_t)M1 * PD * sizeof(unsigned short);   //   368,640
  const size_t SZ_W  = (size_t)PD * PD2 * sizeof(unsigned short);  // 47,775,744
  const size_t SZ_T1 = (size_t)M1 * PD2 * sizeof(unsigned short);  // 106,168,320
  size_t off = 0;
  unsigned short* L1b = (unsigned short*)(ws + off); off += SZ_L;
  unsigned short* L2b = (unsigned short*)(ws + off); off += SZ_L;
  unsigned short* L3b = (unsigned short*)(ws + off); off += SZ_L;
  unsigned short* WbT = (unsigned short*)(ws + off); off += SZ_W;
  unsigned short* T1  = (unsigned short*)(ws + off); off += SZ_T1;
  if (off > ws_size) return;  // workspace too small; cannot run safely

  const int padBlocks = (M1 * PD + 255) / 256;                 // 720
  pad_bias_bf16<<<padBlocks, 256, 0, stream>>>(l1, L1b);
  pad_bias_bf16<<<padBlocks, 256, 0, stream>>>(l2, L2b);
  pad_bias_bf16<<<padBlocks, 256, 0, stream>>>(l3, L3b);
  w_transpose_bf16<<<(PD * PD2 + 255) / 256, 256, 0, stream>>>(W, WbT);

  // stage 1: 640 x 82944 GEMM, K=288. WG tile 64x256 (8 waves, 32x64 each).
  triaffine_stage1<<<dim3(PD2 / 256, M1 / 64, 1), 256, 0, stream>>>(L1b, WbT, T1);

  // stages 2+3 fused: one WG per (n,i), P lives in LDS (160*296*2 = 94,720 B).
  triaffine_stage23<<<dim3(M1, 1, 1), 256, (size_t)LEN * LDSTRIDE * 2, stream>>>(
      L2b, L3b, T1, out);
}
